// CalScaleOPTAttention_59012850647544
// MI455X (gfx1250) — 16-bit and 8-bit matrix operand order repaired and checked by reading, compiled, not run on hardware
//
#include <hip/hip_runtime.h>
#include <hip/hip_bf16.h>
#include <math.h>
#include <stdint.h>

typedef __attribute__((ext_vector_type(16))) _Float16 v16h;
typedef __attribute__((ext_vector_type(8)))  float    v8f;
typedef __attribute__((ext_vector_type(8)))  int      v8i;

#define EPSQ 1e-5f

__device__ inline uint32_t lds_addr_of(const void* p) {
  return (uint32_t)(uintptr_t)(__attribute__((address_space(3))) const char*)p;
}

__device__ inline void async_copy_b128(uint32_t lds_off, const void* gptr) {
  asm volatile("global_load_async_to_lds_b128 %0, %1, off"
               :: "v"(lds_off), "v"(gptr) : "memory");
}
__device__ inline void wait_async0() {
  asm volatile("s_wait_asynccnt 0x0" ::: "memory");
}

__global__ __launch_bounds__(256) void quant_rows_i8(
    const float* __restrict__ x, int8_t* __restrict__ q,
    float* __restrict__ scale, int ncols) {
  int row = blockIdx.x;
  const float* xr = x + (size_t)row * ncols;
  __shared__ float red[256];
  float m = 0.f;
  for (int c = threadIdx.x; c < ncols; c += 256) m = fmaxf(m, fabsf(xr[c]));
  red[threadIdx.x] = m;
  __syncthreads();
  for (int s = 128; s > 0; s >>= 1) {
    if (threadIdx.x < s) red[threadIdx.x] = fmaxf(red[threadIdx.x], red[threadIdx.x + s]);
    __syncthreads();
  }
  float sc = fmaxf(red[0], EPSQ) / 127.f;
  if (threadIdx.x == 0) scale[row] = sc;
  for (int c = threadIdx.x; c < ncols; c += 256)
    q[(size_t)row * ncols + c] = (int8_t)__float2int_rn(xr[c] / sc);
}

__global__ __launch_bounds__(256) void fakequant_rows_q8(float* x, int ncols) {
  int row = blockIdx.x;
  float* xr = x + (size_t)row * ncols;
  __shared__ float red[256];
  float m = 0.f;
  for (int c = threadIdx.x; c < ncols; c += 256) m = fmaxf(m, fabsf(xr[c]));
  red[threadIdx.x] = m;
  __syncthreads();
  for (int s = 128; s > 0; s >>= 1) {
    if (threadIdx.x < s) red[threadIdx.x] = fmaxf(red[threadIdx.x], red[threadIdx.x + s]);
    __syncthreads();
  }
  float sc = fmaxf(red[0], EPSQ) / 127.f;
  for (int c = threadIdx.x; c < ncols; c += 256) xr[c] = rintf(xr[c] / sc) * sc;
}

#define GBM 64
#define GBN 64
#define GKC 128

__global__ __launch_bounds__(256) void gemm_i8_wmma(
    const int8_t* __restrict__ qx, const float* __restrict__ sx,
    const int8_t* __restrict__ qw, const float* __restrict__ sw,
    const float* __restrict__ bias, float* __restrict__ out,
    int M, int N, int K, float mulscale) {
  extern __shared__ char smem[];
  const uint32_t ldsBase = lds_addr_of(smem);

  int tilesN = N / GBN;
  int bm0 = (blockIdx.x / tilesN) * GBM;
  int bn0 = (blockIdx.x % tilesN) * GBN;
  int tid = threadIdx.x;
  int wid = tid >> 5, lane = tid & 31;
  int hh = lane >> 4, l16 = lane & 15;
  int wn = wid & 3;
  int mt0 = (wid >> 2) * 2, mt1 = mt0 + 1;

  int c0 = tid * 2;
  int rowA0 = c0 >> 3, segA0 = (c0 & 7) * 16;
  int rowA1 = (c0 + 1) >> 3, segA1 = ((c0 + 1) & 7) * 16;
  const int8_t* gA = qx + (size_t)(bm0)*K;
  const int8_t* gB = qw + (size_t)(bn0)*K;

  v8i acc0 = {}, acc1 = {};
  int nchunks = K / GKC;

  {
    uint32_t Aoff = ldsBase;
    uint32_t Boff = ldsBase + 2 * (GBM * GKC);
    async_copy_b128(Aoff + rowA0 * GKC + segA0, gA + (size_t)rowA0 * K + segA0);
    async_copy_b128(Aoff + rowA1 * GKC + segA1, gA + (size_t)rowA1 * K + segA1);
    async_copy_b128(Boff + rowA0 * GKC + segA0, gB + (size_t)rowA0 * K + segA0);
    async_copy_b128(Boff + rowA1 * GKC + segA1, gB + (size_t)rowA1 * K + segA1);
  }
  for (int c = 0; c < nchunks; ++c) {
    wait_async0();
    __syncthreads();
    if (c + 1 < nchunks) {
      int k0 = (c + 1) * GKC;
      int nb = (c + 1) & 1;
      uint32_t Aoff = ldsBase + (uint32_t)nb * (GBM * GKC);
      uint32_t Boff = ldsBase + 2 * (GBM * GKC) + (uint32_t)nb * (GBN * GKC);
      async_copy_b128(Aoff + rowA0 * GKC + segA0, gA + (size_t)rowA0 * K + k0 + segA0);
      async_copy_b128(Aoff + rowA1 * GKC + segA1, gA + (size_t)rowA1 * K + k0 + segA1);
      async_copy_b128(Boff + rowA0 * GKC + segA0, gB + (size_t)rowA0 * K + k0 + segA0);
      async_copy_b128(Boff + rowA1 * GKC + segA1, gB + (size_t)rowA1 * K + k0 + segA1);
    }
    const char* Ac = smem + (c & 1) * (GBM * GKC);
    const char* Bc = smem + 2 * (GBM * GKC) + (c & 1) * (GBN * GKC);
    const char* Ar0 = Ac + (mt0 * 16 + l16) * GKC;
    const char* Ar1 = Ac + (mt1 * 16 + l16) * GKC;
    const char* Br  = Bc + (wn * 16 + l16) * GKC;
#pragma unroll
    for (int kk = 0; kk < GKC; kk += 64) {
      v8i a0, a1, b;
#pragma unroll
      for (int v = 0; v < 8; ++v) {
        int ka = kk + ((v >> 1) << 4) + (hh << 3) + ((v & 1) << 2);
        a0[v] = *(const int*)(Ar0 + ka);
        a1[v] = *(const int*)(Ar1 + ka);
        int kb = kk + ((v >> 1) << 4) + (hh << 3) + ((v & 1) << 2);
        b[v] = *(const int*)(Br + kb);
      }
      acc0 = __builtin_amdgcn_wmma_i32_16x16x64_iu8(true, a0, true, b, acc0, false, false);
      acc1 = __builtin_amdgcn_wmma_i32_16x16x64_iu8(true, a1, true, b, acc1, false, false);
    }
    __syncthreads();
  }
  int n = bn0 + wn * 16 + l16;
  float swn = sw[n], bn = bias[n];
#pragma unroll
  for (int r = 0; r < 8; ++r) {
    int mr = bm0 + mt0 * 16 + r + (hh << 3);
    out[(size_t)mr * N + n] = ((float)acc0[r] * sx[mr] * swn + bn) * mulscale;
  }
#pragma unroll
  for (int r = 0; r < 8; ++r) {
    int mr = bm0 + mt1 * 16 + r + (hh << 3);
    out[(size_t)mr * N + n] = ((float)acc1[r] * sx[mr] * swn + bn) * mulscale;
  }
}

__global__ __launch_bounds__(256) void split_heads_f16(
    const float* __restrict__ x, _Float16* __restrict__ out, int T, int H, int D) {
  size_t i = (size_t)blockIdx.x * 256 + threadIdx.x;
  int E = H * D;
  if (i >= (size_t)T * E) return;
  int t = (int)(i / E), e = (int)(i % E);
  int hh = e / D, d = e % D;
  out[((size_t)hh * T + t) * D + d] = (_Float16)x[i];
}

__global__ __launch_bounds__(256) void attn_pass1(
    const _Float16* __restrict__ qh, const _Float16* __restrict__ kh,
    const float* __restrict__ mask, float* __restrict__ accg, int T, int D) {
  extern __shared__ char smem[];
  float* logits = (float*)smem;
  __shared__ float rowred[16][17];
  int hidx = blockIdx.y;
  int t0 = blockIdx.x * 16;
  const _Float16* qb = qh + ((size_t)hidx * T + t0) * D;
  const _Float16* kb = kh + (size_t)hidx * T * D;
  int wid = threadIdx.x >> 5, lane = threadIdx.x & 31;
  int hh = lane >> 4, l16 = lane & 15;
  int ntiles = T >> 4;
  for (int st = wid; st < ntiles; st += 8) {
    int s0 = st << 4;
    v8f acc = {};
    for (int d0 = 0; d0 < D; d0 += 32) {
      v16h a, b;
      const _Float16* qr = qb + (size_t)l16 * D + d0;
#pragma unroll
      for (int v = 0; v < 8; ++v) {
        int ka = ((v >> 2) << 4) + (hh << 3) + ((v & 3) << 1);
        a[2 * v] = qr[ka];
        a[2 * v + 1] = qr[ka + 1];
      }
      const _Float16* kr = kb + (size_t)(s0 + l16) * D + d0 + (hh << 3);
#pragma unroll
      for (int j = 0; j < 16; ++j) b[j] = kr[(j & 7) + ((j >> 3) << 4)];
      acc = __builtin_amdgcn_wmma_f32_16x16x32_f16(false, a, false, b, (short)0,
                                                   acc, false, false);
    }
#pragma unroll
    for (int r = 0; r < 8; ++r) {
      int mrow = r + (hh << 3);
      int scol = s0 + l16;
      logits[mrow * T + scol] = acc[r] + mask[(size_t)(t0 + mrow) * T + scol];
    }
  }
  __syncthreads();
  int row = threadIdx.x >> 4, sub = threadIdx.x & 15;
  float mx = -3.4028235e38f;
  for (int s = sub; s < T; s += 16) mx = fmaxf(mx, logits[row * T + s]);
  rowred[row][sub] = mx;
  __syncthreads();
  if (sub == 0) {
    float v2 = rowred[row][0];
    for (int i = 1; i < 16; ++i) v2 = fmaxf(v2, rowred[row][i]);
    rowred[row][16] = v2;
  }
  __syncthreads();
  mx = rowred[row][16];
  __syncthreads();
  float sum = 0.f;
  for (int s = sub; s < T; s += 16) {
    float p = __expf(logits[row * T + s] - mx);
    logits[row * T + s] = p;
    sum += p;
  }
  rowred[row][sub] = sum;
  __syncthreads();
  if (sub == 0) {
    float v2 = 0.f;
    for (int i = 0; i < 16; ++i) v2 += rowred[row][i];
    rowred[row][16] = v2;
  }
  __syncthreads();
  float inv = 1.f / rowred[row][16];
  for (int s = sub; s < T; s += 16) logits[row * T + s] *= inv;
  __syncthreads();
  for (int s = threadIdx.x; s < T; s += 256) {
    float c = 0.f;
#pragma unroll
    for (int r = 0; r < 16; ++r) c += logits[r * T + s];
    atomicAdd(accg + s, c);
  }
}

__global__ __launch_bounds__(256) void topk_kernel(float* acc, int* istop,
                                                   int T, int BH) {
  for (int s = threadIdx.x; s < T; s += 256) {
    acc[s] = acc[s] / (float)(T - s) / (float)BH;
    istop[s] = 0;
  }
  __syncthreads();
  __shared__ float bm[256];
  __shared__ int bi[256];
  int ktop = T / 40;
  int seglen = T / 4;
  for (int seg = 0; seg < 4; ++seg) {
    int st = seg * seglen, en = st + seglen;
    for (int it = 0; it < ktop; ++it) {
      float best = -1e30f;
      int bidx = st;
      for (int s = st + threadIdx.x; s < en; s += 256) {
        if (!istop[s] && acc[s] > best) { best = acc[s]; bidx = s; }
      }
      bm[threadIdx.x] = best; bi[threadIdx.x] = bidx;
      __syncthreads();
      if (threadIdx.x == 0) {
        float b = bm[0]; int bx = bi[0];
        for (int i = 1; i < 256; ++i)
          if (bm[i] > b) { b = bm[i]; bx = bi[i]; }
        istop[bx] = 1;
      }
      __syncthreads();
    }
  }
}

__global__ __launch_bounds__(256) void kmix_quant(
    const float* __restrict__ k, const int* __restrict__ istop,
    _Float16* __restrict__ kh, int T, int H, int D) {
  int t = blockIdx.x;
  int E = H * D;
  const float* kr = k + (size_t)t * E;
  __shared__ float red[256];
  float m = 0.f;
  for (int e = threadIdx.x; e < E; e += 256) m = fmaxf(m, fabsf(kr[e]));
  red[threadIdx.x] = m;
  __syncthreads();
  for (int s = 128; s > 0; s >>= 1) {
    if (threadIdx.x < s) red[threadIdx.x] = fmaxf(red[threadIdx.x], red[threadIdx.x + s]);
    __syncthreads();
  }
  float am = red[0];
  float s8 = fmaxf(am / 127.f, EPSQ);
  float s4 = fmaxf(am, EPSQ) / 7.f;
  float sc = istop[t] ? s8 : s4;
  for (int e = threadIdx.x; e < E; e += 256) {
    float q = rintf(kr[e] / sc) * sc;
    kh[((size_t)(e / D) * T + t) * D + (e % D)] = (_Float16)q;
  }
}

__global__ __launch_bounds__(256) void vquant_cols(
    const float* __restrict__ v, _Float16* __restrict__ vhT, int T, int H, int D) {
  int E = H * D;
  int col = threadIdx.x & 63;
  int rgrp = threadIdx.x >> 6;
  int e = blockIdx.x * 64 + col;
  float m = 0.f;
  for (int t = rgrp; t < T; t += 4) m = fmaxf(m, fabsf(v[(size_t)t * E + e]));
  __shared__ float red[4][64];
  red[rgrp][col] = m;
  __syncthreads();
  float am = fmaxf(fmaxf(red[0][col], red[1][col]), fmaxf(red[2][col], red[3][col]));
  float sc = fmaxf(am, EPSQ) / 127.f;
  int hhh = e / D, d = e % D;
  _Float16* dst = vhT + ((size_t)hhh * D + d) * T;
  for (int t = rgrp; t < T; t += 4)
    dst[t] = (_Float16)(rintf(v[(size_t)t * E + e] / sc) * sc);
}

__global__ __launch_bounds__(256) void attn_pass2(
    const _Float16* __restrict__ qh, const _Float16* __restrict__ kh,
    const _Float16* __restrict__ vhT, const float* __restrict__ mask,
    float* __restrict__ ctx, int T, int D, int H) {
  extern __shared__ char smem[];
  float* logits = (float*)smem;
  _Float16* probsH = (_Float16*)(smem + (size_t)16 * T * 4);
  __shared__ float rowred[16][17];
  int hidx = blockIdx.y;
  int t0 = blockIdx.x * 16;
  const _Float16* qb = qh + ((size_t)hidx * T + t0) * D;
  const _Float16* kb = kh + (size_t)hidx * T * D;
  int wid = threadIdx.x >> 5, lane = threadIdx.x & 31;
  int hh = lane >> 4, l16 = lane & 15;
  int ntiles = T >> 4;
  for (int st = wid; st < ntiles; st += 8) {
    int s0 = st << 4;
    v8f acc = {};
    for (int d0 = 0; d0 < D; d0 += 32) {
      v16h a, b;
      const _Float16* qr = qb + (size_t)l16 * D + d0;
#pragma unroll
      for (int v = 0; v < 8; ++v) {
        int ka = ((v >> 2) << 4) + (hh << 3) + ((v & 3) << 1);
        a[2 * v] = qr[ka];
        a[2 * v + 1] = qr[ka + 1];
      }
      const _Float16* kr = kb + (size_t)(s0 + l16) * D + d0 + (hh << 3);
#pragma unroll
      for (int j = 0; j < 16; ++j) b[j] = kr[(j & 7) + ((j >> 3) << 4)];
      acc = __builtin_amdgcn_wmma_f32_16x16x32_f16(false, a, false, b, (short)0,
                                                   acc, false, false);
    }
#pragma unroll
    for (int r = 0; r < 8; ++r) {
      int mrow = r + (hh << 3);
      int scol = s0 + l16;
      logits[mrow * T + scol] = acc[r] + mask[(size_t)(t0 + mrow) * T + scol];
    }
  }
  __syncthreads();
  int row = threadIdx.x >> 4, sub = threadIdx.x & 15;
  float mx = -3.4028235e38f;
  for (int s = sub; s < T; s += 16) mx = fmaxf(mx, logits[row * T + s]);
  rowred[row][sub] = mx;
  __syncthreads();
  if (sub == 0) {
    float v2 = rowred[row][0];
    for (int i = 1; i < 16; ++i) v2 = fmaxf(v2, rowred[row][i]);
    rowred[row][16] = v2;
  }
  __syncthreads();
  mx = rowred[row][16];
  __syncthreads();
  float sum = 0.f;
  for (int s = sub; s < T; s += 16) {
    float p = __expf(logits[row * T + s] - mx);
    logits[row * T + s] = p;
    sum += p;
  }
  rowred[row][sub] = sum;
  __syncthreads();
  if (sub == 0) {
    float v2 = 0.f;
    for (int i = 0; i < 16; ++i) v2 += rowred[row][i];
    rowred[row][16] = v2;
  }
  __syncthreads();
  float inv = 1.f / rowred[row][16];
  for (int s = sub; s < T; s += 16) {
    float p = logits[row * T + s] * inv;
    probsH[row * T + s] = (_Float16)p;
  }
  __syncthreads();
  int n0 = wid << 4;
  v8f acc = {};
  for (int s0 = 0; s0 < T; s0 += 32) {
    v16h a, b;
    const _Float16* pr = probsH + (size_t)l16 * T + s0;
#pragma unroll
    for (int v = 0; v < 8; ++v) {
      int ka = ((v >> 2) << 4) + (hh << 3) + ((v & 3) << 1);
      a[2 * v] = pr[ka];
      a[2 * v + 1] = pr[ka + 1];
    }
    const _Float16* vr = vhT + ((size_t)hidx * D + n0 + l16) * T + s0 + (hh << 3);
#pragma unroll
    for (int j = 0; j < 16; ++j) b[j] = vr[(j & 7) + ((j >> 3) << 4)];
    acc = __builtin_amdgcn_wmma_f32_16x16x32_f16(false, a, false, b, (short)0,
                                                 acc, false, false);
  }
  int E = H * D;
#pragma unroll
  for (int r = 0; r < 8; ++r) {
    int t = t0 + r + (hh << 3);
    ctx[(size_t)t * E + hidx * D + n0 + l16] = acc[r];
  }
}

extern "C" void kernel_launch(void* const* d_in, const int* in_sizes, int n_in,
                              void* d_out, int out_size, void* d_ws, size_t ws_size,
                              hipStream_t stream) {
  (void)in_sizes; (void)n_in; (void)out_size; (void)ws_size;
  const float* hs   = (const float*)d_in[0];
  const float* mask = (const float*)d_in[1];
  const float* q_b  = (const float*)d_in[3];
  const float* k_b  = (const float*)d_in[5];
  const float* v_b  = (const float*)d_in[7];
  const float* o_b  = (const float*)d_in[9];
  const float* Wsrc[4] = {(const float*)d_in[2], (const float*)d_in[4],
                          (const float*)d_in[6], (const float*)d_in[8]};
  const int T = 2048, E = 2048, H = 16, D = 128;

  char* ws = (char*)d_ws;
  size_t off = 0;
  auto alloc = [&](size_t bytes) -> char* {
    char* p = ws + off;
    off += bytes;
    off = (off + 255) & ~(size_t)255;
    return p;
  };
  int8_t* xq = (int8_t*)alloc((size_t)T * E);
  float*  xs = (float*)alloc((size_t)T * 4);
  int8_t* wq[4]; float* wsc[4];
  for (int i = 0; i < 4; ++i) {
    wq[i]  = (int8_t*)alloc((size_t)E * E);
    wsc[i] = (float*)alloc((size_t)E * 4);
  }
  float* qbuf = (float*)alloc((size_t)T * E * 4);
  float* kbuf = (float*)alloc((size_t)T * E * 4);
  float* vbuf = (float*)alloc((size_t)T * E * 4);
  _Float16* qh  = (_Float16*)alloc((size_t)H * T * D * 2);
  _Float16* khx = (_Float16*)alloc((size_t)H * T * D * 2);
  _Float16* vhT = (_Float16*)alloc((size_t)H * D * T * 2);
  float* accg = (float*)alloc((size_t)T * 4);
  int*   istop = (int*)alloc((size_t)T * 4);
  int8_t* cq = (int8_t*)alloc((size_t)T * E);
  float*  cs = (float*)alloc((size_t)T * 4);

  dim3 b256(256);
  quant_rows_i8<<<T, b256, 0, stream>>>(hs, xq, xs, E);
  for (int i = 0; i < 4; ++i)
    quant_rows_i8<<<E, b256, 0, stream>>>(Wsrc[i], wq[i], wsc[i], E);
  int gemmGrid = (T / GBM) * (E / GBN);
  size_t gemmLds = 2 * (size_t)GBM * GKC + 2 * (size_t)GBN * GKC;
  float scaling = 1.f / sqrtf((float)D);
  gemm_i8_wmma<<<gemmGrid, b256, gemmLds, stream>>>(xq, xs, wq[0], wsc[0], q_b,
                                                    qbuf, T, E, E, scaling);
  gemm_i8_wmma<<<gemmGrid, b256, gemmLds, stream>>>(xq, xs, wq[1], wsc[1], k_b,
                                                    kbuf, T, E, E, 1.0f);
  gemm_i8_wmma<<<gemmGrid, b256, gemmLds, stream>>>(xq, xs, wq[2], wsc[2], v_b,
                                                    vbuf, T, E, E, 1.0f);
  fakequant_rows_q8<<<T, b256, 0, stream>>>(vbuf, E);
  int cvtGrid = (int)(((size_t)T * E + 255) / 256);
  split_heads_f16<<<cvtGrid, b256, 0, stream>>>(qbuf, qh, T, H, D);
  split_heads_f16<<<cvtGrid, b256, 0, stream>>>(kbuf, khx, T, H, D);
  (void)hipMemsetAsync(accg, 0, (size_t)T * 4, stream);
  dim3 agrid(T / 16, H);
  size_t lds1 = (size_t)16 * T * 4;
  attn_pass1<<<agrid, b256, lds1, stream>>>(qh, khx, mask, accg, T, D);
  topk_kernel<<<1, b256, 0, stream>>>(accg, istop, T, H);
  kmix_quant<<<T, b256, 0, stream>>>(kbuf, istop, khx, T, H, D);
  vquant_cols<<<E / 64, b256, 0, stream>>>(vbuf, vhT, T, H, D);
  float* ctx = qbuf;
  size_t lds2 = lds1 + (size_t)16 * T * 2;
  attn_pass2<<<agrid, b256, lds2, stream>>>(qh, khx, vhT, mask, ctx, T, D, H);
  quant_rows_i8<<<T, b256, 0, stream>>>(ctx, cq, cs, E);
  gemm_i8_wmma<<<gemmGrid, b256, gemmLds, stream>>>(cq, cs, wq[3], wsc[3], o_b,
                                                    (float*)d_out, T, E, E, 1.0f);
}
